// MHSelfAttention_39651138076975
// MI455X (gfx1250) — matrix operand order repaired and checked by reading, not compiled, not run on hardware
//
#include <hip/hip_runtime.h>

// ---------------------------------------------------------------------------
// MI455X / gfx1250 causal multi-head self-attention, bf16 WMMA + async/TDM.
// ---------------------------------------------------------------------------

typedef __attribute__((ext_vector_type(16))) __bf16 v16bf;
typedef __attribute__((ext_vector_type(8)))  float  v8f;
typedef __attribute__((ext_vector_type(4)))  unsigned int u32x4;
typedef __attribute__((ext_vector_type(8)))  int i32x8;
typedef __attribute__((ext_vector_type(4)))  int i32x4;

#if defined(__has_builtin)
#if __has_builtin(__builtin_amdgcn_tensor_load_to_lds)
#define HAVE_TDM 1
#endif
#endif
#ifndef HAVE_TDM
#define HAVE_TDM 0
#endif

union FragU { v16bf v; unsigned int u[8]; };

__device__ __forceinline__ unsigned short f2bf(float f) {
  unsigned int u = __float_as_uint(f);
  u += 0x7FFFu + ((u >> 16) & 1u);   // round-to-nearest-even
  return (unsigned short)(u >> 16);
}

__device__ __forceinline__ v8f wmma_bf16(v16bf a, v16bf b, v8f c) {
  return __builtin_amdgcn_wmma_f32_16x16x32_bf16(false, a, false, b,
                                                 (short)0, c, false, false);
}

// LDS byte offset of a __shared__ object (generic LDS addr keeps offset in [31:0])
__device__ __forceinline__ unsigned lds_off(const void* p) {
  return (unsigned)(unsigned long long)p;
}

// Per-lane async 16B global->LDS copy (ASYNCcnt).
__device__ __forceinline__ void async_b128(unsigned lds, const void* g) {
  asm volatile("global_load_async_to_lds_b128 %0, %1, off"
               :: "v"(lds), "v"(g) : "memory");
}

// A-fragment 16x32 bf16 from row-major LDS [m][k].
// Contiguous K: VGPR v holds K=16*half+2v,+1 ; lanes 0-15/16-31 -> M.
__device__ __forceinline__ v16bf ld_frag_a(const unsigned short* p, int stride,
                                           int koff, int lane) {
  const int idx = lane & 15, half = lane >> 4;
  FragU f;
  const unsigned short* row = p + idx * stride + koff;
#pragma unroll
  for (int v = 0; v < 8; ++v) {
    int kb = 2 * v + 16 * half;
    f.u[v] = *(const unsigned int*)(row + kb);
  }
  return f.v;
}

// B-fragment 32x16 bf16 from LDS stored [n][k] (contiguous K per row).
__device__ __forceinline__ v16bf ld_frag_b_nk(const unsigned short* p, int stride,
                                              int koff, int lane) {
  const int idx = lane & 15, half = lane >> 4;
  FragU f;
  const unsigned short* row = p + idx * stride + koff + 16 * half;
#pragma unroll
  for (int v = 0; v < 8; ++v) f.u[v] = *(const unsigned int*)(row + 2 * v);
  return f.v;
}

// B-fragment 32x16 bf16 from LDS stored [k][n] (no staging transpose needed):
// lane -> column n, gather 16 K values with stride, pack pairs into dwords.
__device__ __forceinline__ v16bf ld_frag_b_kn(const unsigned short* p, int stride,
                                              int koff, int lane) {
  const int n = lane & 15, half = lane >> 4;
  FragU f;
  const unsigned short* base = p + (koff + 16 * half) * stride + n;
#pragma unroll
  for (int v = 0; v < 8; ++v) {
    unsigned lo = base[(2 * v) * stride];
    unsigned hi = base[(2 * v + 1) * stride];
    f.u[v] = lo | (hi << 16);
  }
  return f.v;
}

#if HAVE_TDM
// TDM: contiguous 1-tile bf16 copy global -> LDS (D# per CDNA5 ISA ch.8).
__device__ __forceinline__ void tdm_copy_1d(unsigned lds, const void* gaddr,
                                            unsigned elems) {
  unsigned long long ga = (unsigned long long)gaddr;
  u32x4 g0 = {1u,                                   // count=1 (valid user D#)
              lds,                                  // lds_addr
              (unsigned)(ga & 0xFFFFFFFFu),         // global_addr[31:0]
              (unsigned)((ga >> 32) & 0x01FFFFFFu) | (2u << 30)}; // [56:32]|type=2
  i32x8 g1 = {(int)(1u << 16),                      // data_size=1 (2B), mask=0
              (int)((elems & 0xFFFFu) << 16),       // tensor_dim0[15:0]
              (int)(((elems >> 16) & 0xFFFFu) | (1u << 16)), // dim0[31:16]|dim1=1
              (int)((elems & 0xFFFFu) << 16),       // tile_dim0
              1,                                    // tile_dim1=1
              (int)elems,                           // tensor_dim0_stride[31:0]
              0, 0};
  i32x4 z4 = {0, 0, 0, 0};
#if defined(__clang_major__) && (__clang_major__ >= 23)
  i32x8 z8 = {0, 0, 0, 0, 0, 0, 0, 0};
  __builtin_amdgcn_tensor_load_to_lds(g0, g1, z4, z4, z8, 0);
#else
  __builtin_amdgcn_tensor_load_to_lds(g0, g1, z4, z4, 0);
#endif
}
#endif

// ---------------------------------------------------------------------------
// Kernel 0: fp32 -> bf16 conversion
// ---------------------------------------------------------------------------
__global__ void cvt_f32_bf16(const float* __restrict__ src,
                             unsigned short* __restrict__ dst, int n) {
  int i = blockIdx.x * blockDim.x + threadIdx.x;
  if (i < n) dst[i] = f2bf(src[i]);
}

// ---------------------------------------------------------------------------
// Kernel 1/3: bf16 WMMA GEMM  C[M,N] = A[M,K]*B[K,N] + bias, async-pipelined.
// Block tile 128x128, BK=32, 256 threads = 8 waves (2x4), wave tile 64x32.
//   mode 0: scatter Q/K/V bf16 [B][H][S][Hd] ; mode 1: fp32 row-major out.
// ---------------------------------------------------------------------------
#define BM 128
#define BN 128
#define BK 32

__global__ __launch_bounds__(256) void gemm_bf16(
    const unsigned short* __restrict__ A, const unsigned short* __restrict__ Bw,
    const float* __restrict__ bias, int M, int N, int K, int mode,
    unsigned short* __restrict__ q_out, unsigned short* __restrict__ k_out,
    unsigned short* __restrict__ v_out, float* __restrict__ c_out) {
  __shared__ __align__(16) unsigned short As[2][BM][BK];   // [m][k]
  __shared__ __align__(16) unsigned short Bs[2][BK][BN];   // [k][n] (no transpose)

  const int tid = threadIdx.x, lane = tid & 31, wave = tid >> 5;
  const int wm = wave >> 2, wn = wave & 3;
  const int m0 = blockIdx.y * BM, n0 = blockIdx.x * BN;

  const v8f zf = {0.f, 0.f, 0.f, 0.f, 0.f, 0.f, 0.f, 0.f};
  v8f acc[4][2];
#pragma unroll
  for (int i = 0; i < 4; ++i)
#pragma unroll
    for (int j = 0; j < 2; ++j) acc[i][j] = zf;

  // Per-thread async staging of one BMxBK + BKxBN tile pair (4 b128 ops).
  auto stage = [&](int t, int buf) {
    const int k0 = t * BK;
#pragma unroll
    for (int i = 0; i < 2; ++i) {
      int seg = tid + i * 256;                  // A: 512 x 8 elements
      int r = seg >> 2, c = (seg & 3) * 8;
      async_b128(lds_off(&As[buf][r][c]),
                 &A[(size_t)(m0 + r) * K + k0 + c]);
    }
#pragma unroll
    for (int i = 0; i < 2; ++i) {
      int seg = tid + i * 256;                  // B: straight [k][n] copy
      int kr = seg >> 4, ns = (seg & 15) * 8;
      async_b128(lds_off(&Bs[buf][kr][ns]),
                 &Bw[(size_t)(k0 + kr) * N + n0 + ns]);
    }
  };

  const int nk = K / BK;
  stage(0, 0);
  for (int t = 0; t < nk; ++t) {
    const int buf = t & 1;
    const bool more = (t + 1 < nk);
    if (more) stage(t + 1, buf ^ 1);
    // in-order async completion: <=4 outstanding means tile t has landed
    if (more) asm volatile("s_wait_asynccnt 0x4" ::: "memory");
    else      asm volatile("s_wait_asynccnt 0x0" ::: "memory");
    __syncthreads();

    v16bf bfr[2];
#pragma unroll
    for (int nt = 0; nt < 2; ++nt)
      bfr[nt] = ld_frag_b_kn(&Bs[buf][0][wn * 32 + nt * 16], BN, 0, lane);
#pragma unroll
    for (int mt = 0; mt < 4; ++mt) {
      v16bf af = ld_frag_a(&As[buf][wm * 64 + mt * 16][0], BK, 0, lane);
#pragma unroll
      for (int nt = 0; nt < 2; ++nt)
        acc[mt][nt] = wmma_bf16(af, bfr[nt], acc[mt][nt]);
    }
    __syncthreads();
  }

  // ---- epilogue (C/D: lane -> N, vgpr r -> M = r + 8*half) ----
  const int nlo = lane & 15, half = lane >> 4;
#pragma unroll
  for (int mt = 0; mt < 4; ++mt)
#pragma unroll
    for (int nt = 0; nt < 2; ++nt)
#pragma unroll
      for (int r = 0; r < 8; ++r) {
        int m = m0 + wm * 64 + mt * 16 + r + 8 * half;
        int n = n0 + wn * 32 + nt * 16 + nlo;
        float val = acc[mt][nt][r] + bias[n];
        if (mode == 0) {
          int which = n >> 10, h = (n >> 6) & 15, d = n & 63;
          int b = m >> 11, s = m & 2047;
          unsigned short* dst =
              (which == 0) ? q_out : ((which == 1) ? k_out : v_out);
          dst[((size_t)((b * 16 + h) * 2048 + s)) * 64 + d] = f2bf(val);
        } else {
          c_out[(size_t)m * N + n] = val;
        }
      }
}

// ---------------------------------------------------------------------------
// Kernel 2: causal flash attention.
// Grid: x = q-block (32), y = b*H (32). Block: 128 threads = 4 waves,
// each wave owns 16 query rows. KV tiles (64 keys) staged by the Tensor Data
// Mover (TENSORcnt) when available, else async-to-LDS; double buffered.
// ---------------------------------------------------------------------------
__global__ __launch_bounds__(128) void attn_flash(
    const unsigned short* __restrict__ Qb, const unsigned short* __restrict__ Kb,
    const unsigned short* __restrict__ Vb, unsigned short* __restrict__ attn) {
  __shared__ __align__(16) unsigned short Qs[64 * 64];      // [q][d]
  __shared__ __align__(16) unsigned short Ks[2][64 * 64];   // [key][d]
  __shared__ __align__(16) unsigned short Vs[2][64 * 64];   // [key][d]
  __shared__ __align__(16) unsigned short Ps[4][16 * 64];   // per-wave P [q][key]

  const int tid = threadIdx.x, lane = tid & 31, wave = tid >> 5;
  const int qb = blockIdx.x;
  const int bh = blockIdx.y;
  const int nlo = lane & 15, half = lane >> 4;
  const size_t head_off = (size_t)bh * 2048 * 64;

  const unsigned short* Qg = Qb + head_off + (size_t)qb * 64 * 64;
  const unsigned short* Kh = Kb + head_off;
  const unsigned short* Vh = Vb + head_off;

  // Stage K/V tile j into buffer buf.
  auto stage_kv = [&](int j, int buf) {
#if HAVE_TDM
    if (wave == 0) {
      tdm_copy_1d(lds_off(&Ks[buf][0]), Kh + (size_t)j * 4096, 4096);
      tdm_copy_1d(lds_off(&Vs[buf][0]), Vh + (size_t)j * 4096, 4096);
    }
#else
#pragma unroll
    for (int i = 0; i < 4; ++i) {
      int seg = tid + i * 128;
      async_b128(lds_off(&Ks[buf][seg * 8]), Kh + (size_t)j * 4096 + seg * 8);
    }
#pragma unroll
    for (int i = 0; i < 4; ++i) {
      int seg = tid + i * 128;
      async_b128(lds_off(&Vs[buf][seg * 8]), Vh + (size_t)j * 4096 + seg * 8);
    }
#endif
  };
  auto wait_kv = [&](bool more) {
#if HAVE_TDM
    if (wave == 0) {
      if (more) __builtin_amdgcn_s_wait_tensorcnt(2);
      else      __builtin_amdgcn_s_wait_tensorcnt(0);
    }
#else
    if (more) asm volatile("s_wait_asynccnt 0x8" ::: "memory");
    else      asm volatile("s_wait_asynccnt 0x0" ::: "memory");
#endif
    __syncthreads();
  };

  // ---- load Q block (once) ----
#pragma unroll
  for (int i = 0; i < 4; ++i) {
    int seg = tid + i * 128;
    *(uint4*)&Qs[seg * 8] = *(const uint4*)&Qg[seg * 8];
  }
  stage_kv(0, 0);
  __syncthreads();

  v16bf qf0 = ld_frag_a(&Qs[(wave * 16) * 64], 64, 0, lane);
  v16bf qf1 = ld_frag_a(&Qs[(wave * 16) * 64], 64, 32, lane);

  const v8f zf = {0.f, 0.f, 0.f, 0.f, 0.f, 0.f, 0.f, 0.f};
  v8f o[4];
#pragma unroll
  for (int dt = 0; dt < 4; ++dt) o[dt] = zf;
  float mi[8], li[8];
#pragma unroll
  for (int r = 0; r < 8; ++r) { mi[r] = -1e30f; li[r] = 0.f; }

  for (int j = 0; j <= qb; ++j) {        // causal: KV blocks <= diagonal
    const int buf = j & 1;
    const bool more = (j < qb);
    if (more) stage_kv(j + 1, buf ^ 1);
    wait_kv(more);                       // tile j resident + barrier

    // ---- S = (Q K^T)/8 : 4 key-tiles x 2 d-steps ----
    v8f sacc[4];
#pragma unroll
    for (int nt = 0; nt < 4; ++nt) {
      v8f s = zf;
      v16bf kb0 = ld_frag_b_nk(&Ks[buf][(nt * 16) * 64], 64, 0, lane);
      s = wmma_bf16(qf0, kb0, s);
      v16bf kb1 = ld_frag_b_nk(&Ks[buf][(nt * 16) * 64], 64, 32, lane);
      s = wmma_bf16(qf1, kb1, s);
      sacc[nt] = s;
    }

    const bool diag = (j == qb);
    float mnew[8];
#pragma unroll
    for (int r = 0; r < 8; ++r) {
      int m = r + 8 * half;
      float mx = mi[r];
#pragma unroll
      for (int nt = 0; nt < 4; ++nt) {
        float sv = sacc[nt][r] * 0.125f;
        if (diag && (nt * 16 + nlo) > (wave * 16 + m)) sv = -1e30f;
        sacc[nt][r] = sv;
        mx = fmaxf(mx, sv);
      }
#pragma unroll
      for (int msk = 1; msk <= 8; msk <<= 1)
        mx = fmaxf(mx, __shfl_xor(mx, msk, 32));
      mnew[r] = mx;
    }

    // ---- P = exp(S - mnew); stash bf16 P in per-wave LDS; row sums ----
    unsigned short* Pw = &Ps[wave][0];
    float lsum[8];
#pragma unroll
    for (int r = 0; r < 8; ++r) {
      int m = r + 8 * half;
      float sum = 0.f;
#pragma unroll
      for (int nt = 0; nt < 4; ++nt) {
        float p = __expf(sacc[nt][r] - mnew[r]);
        sum += p;
        Pw[m * 64 + nt * 16 + nlo] = f2bf(p);
      }
#pragma unroll
      for (int msk = 1; msk <= 8; msk <<= 1) sum += __shfl_xor(sum, msk, 32);
      lsum[r] = sum;
    }

    // ---- rescale running state ----
#pragma unroll
    for (int r = 0; r < 8; ++r) {
      float alpha = __expf(mi[r] - mnew[r]);
      li[r] = li[r] * alpha + lsum[r];
      mi[r] = mnew[r];
#pragma unroll
      for (int dt = 0; dt < 4; ++dt) o[dt][r] *= alpha;
    }

    // cross-lane LDS round trip within this wave: drain DS before reload
    asm volatile("s_wait_dscnt 0" ::: "memory");

    // ---- O += P V : 2 key-steps x 4 d-tiles (V gathered from [key][d]) ----
    v16bf pf0 = ld_frag_a(Pw, 64, 0, lane);
    v16bf pf1 = ld_frag_a(Pw, 64, 32, lane);
#pragma unroll
    for (int dt = 0; dt < 4; ++dt) {
      v16bf vb0 = ld_frag_b_kn(&Vs[buf][dt * 16], 64, 0, lane);
      o[dt] = wmma_bf16(pf0, vb0, o[dt]);
      v16bf vb1 = ld_frag_b_kn(&Vs[buf][dt * 16], 64, 32, lane);
      o[dt] = wmma_bf16(pf1, vb1, o[dt]);
    }
    __syncthreads();                     // release buf before it is restaged
  }

  // ---- epilogue: normalize, store bf16 attn as [B][S][H*64] ----
  const int b = bh >> 4, h = bh & 15;
#pragma unroll
  for (int dt = 0; dt < 4; ++dt)
#pragma unroll
    for (int r = 0; r < 8; ++r) {
      int m = r + 8 * half;
      int row = qb * 64 + wave * 16 + m;
      float val = o[dt][r] / li[r];
      attn[((size_t)(b * 2048 + row)) * 1024 + h * 64 + dt * 16 + nlo] =
          f2bf(val);
    }
}

// ---------------------------------------------------------------------------
// Host-side launcher
// ---------------------------------------------------------------------------
extern "C" void kernel_launch(void* const* d_in, const int* in_sizes, int n_in,
                              void* d_out, int out_size, void* d_ws,
                              size_t ws_size, hipStream_t stream) {
  const float* X    = (const float*)d_in[0];  // [2,2048,1024]
  const float* Wqkv = (const float*)d_in[1];  // [1024,3072]
  const float* bqkv = (const float*)d_in[2];  // [3072]
  const float* Wout = (const float*)d_in[3];  // [1024,1024]
  const float* bout = (const float*)d_in[4];  // [1024]
  float* out = (float*)d_out;

  const int NX  = 2 * 2048 * 1024;
  const int NW1 = 1024 * 3072;
  const int NW2 = 1024 * 1024;

  unsigned short* ws = (unsigned short*)d_ws;
  unsigned short* Xbf    = ws;
  unsigned short* Wqkvbf = Xbf + NX;
  unsigned short* Woutbf = Wqkvbf + NW1;
  unsigned short* Qb     = Woutbf + NW2;
  unsigned short* Kb     = Qb + NX;
  unsigned short* Vb     = Kb + NX;
  unsigned short* Attnbf = Vb + NX;

  cvt_f32_bf16<<<NX / 256, 256, 0, stream>>>(X, Xbf, NX);
  cvt_f32_bf16<<<NW1 / 256, 256, 0, stream>>>(Wqkv, Wqkvbf, NW1);
  cvt_f32_bf16<<<NW2 / 256, 256, 0, stream>>>(Wout, Woutbf, NW2);

  gemm_bf16<<<dim3(3072 / BN, 4096 / BM), 256, 0, stream>>>(
      Xbf, Wqkvbf, bqkv, 4096, 3072, 1024, 0, Qb, Kb, Vb, nullptr);

  attn_flash<<<dim3(32, 32), 128, 0, stream>>>(Qb, Kb, Vb, Attnbf);

  gemm_bf16<<<dim3(1024 / BN, 4096 / BM), 256, 0, stream>>>(
      Attnbf, Woutbf, bout, 4096, 1024, 1024, 1, nullptr, nullptr, nullptr,
      out);
}
